// AliasFreeActivation_80427557584918
// MI455X (gfx1250) — matrix operand order repaired and checked by reading, not compiled, not run on hardware
//
#include <hip/hip_runtime.h>
#include <math.h>

typedef _Float16 h16;
typedef __attribute__((ext_vector_type(16))) _Float16 v16h;
typedef __attribute__((ext_vector_type(8)))  _Float16 v8h;
typedef __attribute__((ext_vector_type(8)))  float    v8f;
typedef __attribute__((ext_vector_type(4)))  float    v4f;

#define HH 128
#define WW 128
#define CC 256
#define OT 16   // output tile edge
#define PR 26   // input patch (with halo) actual edge
#define PA 32   // padded patch edge (pass1 M and K)
#define IA 48   // padded intermediate edge (42 -> 48)
#define KZ 64   // padded K for down passes (42 -> 64)

#define SHUF16(LO,HI) __builtin_shufflevector(LO,HI,0,1,2,3,4,5,6,7,8,9,10,11,12,13,14,15)
#define WMMA(A,B,C) __builtin_amdgcn_wmma_f32_16x16x32_f16(false,(A),false,(B),(short)0,(C),false,false)

__global__ __launch_bounds__(32)
void flrelu_fused_wmma(const float* __restrict__ x, const float* __restrict__ bias,
                       const float* __restrict__ upf, const float* __restrict__ dnf,
                       float* __restrict__ out)
{
  // Constant banded filter matrices (identical for every tile thanks to 16-alignment)
  __shared__ h16 UU[IA][PA];   // UU[a][b] = u[1 - a + 2b]   (up FIR, A role in pass2, B role in pass1)
  __shared__ h16 DD[OT][KZ];   // DD[a][b] = d[b - 2a]       (down FIR, B role in pass4, A role in pass5)
  __shared__ h16 Xp [PA][PA];  // input patch (bias added, f16), zero halo
  __shared__ h16 V1T[IA][PA];  // pass1 result, stored as B-layout [dx][i] for pass2
  __shared__ h16 ZT [IA][KZ];  // pass2+lrelu result, [dx][dy]; cols 48..63 zeroed
  __shared__ h16 OvT[OT][KZ];  // pass4 result, stored as B-layout [p][dx] for pass5

  const int lane = threadIdx.x;
  const int lln  = lane & 15;
  const int lhi  = lane >> 4;

  const int tix = blockIdx.x & 7;
  const int tiy = blockIdx.x >> 3;
  const int c   = blockIdx.y;
  const int b   = blockIdx.z;
  const int oy0 = tiy * OT, ox0 = tix * OT;
  const int r0  = oy0 - 5,  c0  = ox0 - 5;   // input halo origin

  // Recover separable 1-D factors from the 2-D outer-product filters:
  // f = u (outer) u  =>  u[i] = f[i][5] / sqrt(f[5][5])  (Kaiser taps are positive)
  const float ur = 1.0f / sqrtf(upf[5*12 + 5]);
  const float dr = 1.0f / sqrtf(dnf[5*12 + 5]);

  #pragma unroll 4
  for (int idx = lane; idx < IA*PA; idx += 32) {
    int a = idx >> 5, bb = idx & 31;
    int t = 1 - a + 2*bb;
    UU[a][bb] = (h16)((t >= 0 && t < 12) ? upf[t*12 + 5] * ur : 0.0f);
  }
  #pragma unroll 4
  for (int idx = lane; idx < OT*KZ; idx += 32) {
    int a = idx >> 6, bb = idx & 63;
    int t = bb - 2*a;
    DD[a][bb] = (h16)((t >= 0 && t < 12) ? dnf[t*12 + 5] * dr : 0.0f);
  }
  // zero the K-padding columns consumed by passes 4/5
  for (int idx = lane; idx < IA*16; idx += 32) ZT [idx >> 4][48 + (idx & 15)] = (h16)0.0f;
  for (int idx = lane; idx < OT*16; idx += 32) OvT[idx >> 4][48 + (idx & 15)] = (h16)0.0f;

  // ---- load input patch (coalesced: one 32-wide row per iteration), add bias, f16 ----
  const float bi  = bias[c];
  const float* xp = x + ((size_t)(b*CC + c) * HH) * WW;
  #pragma unroll 4
  for (int i = 0; i < PA; ++i) {
    int rr = r0 + i, cc2 = c0 + lane;
    float v = 0.0f;
    if (i < PR && lane < PR && rr >= 0 && rr < HH && cc2 >= 0 && cc2 < WW)
      v = xp[rr*WW + cc2] + bi;
    Xp[i][lane] = (h16)v;
  }

  // ---- Pass 1: horizontal up-FIR.  V1[i,dx] = sum_j Xp[i,j] * UU[dx][j] ----
  #pragma unroll
  for (int m0 = 0; m0 < PA; m0 += 16) {
    v8h alo = *(const v8h*)&Xp[m0 + lln][8*lhi];
    v8h ahi = *(const v8h*)&Xp[m0 + lln][16 + 8*lhi];
    v16h A = SHUF16(alo, ahi);
    #pragma unroll
    for (int n0 = 0; n0 < IA; n0 += 16) {
      v8h blo = *(const v8h*)&UU[n0 + lln][8*lhi];
      v8h bhi = *(const v8h*)&UU[n0 + lln][16 + 8*lhi];
      v16h B = SHUF16(blo, bhi);
      v8f acc = {};
      acc = WMMA(A, B, acc);
      v8h t;
      #pragma unroll
      for (int r = 0; r < 8; ++r) t[r] = (h16)acc[r];
      *(v8h*)&V1T[n0 + lln][m0 + 8*lhi] = t;        // B-layout for pass 2
    }
  }

  // ---- Pass 2: vertical up-FIR + fused lrelu*sqrt(2) + clamp.  Y[dy,dx] = sum_i UU[dy][i]*V1[i,dx] ----
  #pragma unroll
  for (int m0 = 0; m0 < IA; m0 += 16) {
    v8h alo = *(const v8h*)&UU[m0 + lln][8*lhi];
    v8h ahi = *(const v8h*)&UU[m0 + lln][16 + 8*lhi];
    v16h A = SHUF16(alo, ahi);
    #pragma unroll
    for (int n0 = 0; n0 < IA; n0 += 16) {
      v8h blo = *(const v8h*)&V1T[n0 + lln][8*lhi];
      v8h bhi = *(const v8h*)&V1T[n0 + lln][16 + 8*lhi];
      v16h B = SHUF16(blo, bhi);
      v8f acc = {};
      acc = WMMA(A, B, acc);
      v8h t;
      #pragma unroll
      for (int r = 0; r < 8; ++r) {
        float y = acc[r];
        y = (y >= 0.0f ? y : 0.2f * y) * 1.41421356237f;
        y = fminf(fmaxf(y, -256.0f), 256.0f);
        t[r] = (h16)y;
      }
      *(v8h*)&ZT[n0 + lln][m0 + 8*lhi] = t;         // transposed: A-layout for pass 4
    }
  }

  // ---- Pass 4: vertical down-FIR (stride 2 folded into DD band). Out_vT[dx,p] = sum_dzy ZT[dx][dzy]*DD[p][dzy] ----
  #pragma unroll
  for (int m0 = 0; m0 < IA; m0 += 16) {
    v8f acc = {};
    #pragma unroll
    for (int k0 = 0; k0 < KZ; k0 += 32) {
      v8h alo = *(const v8h*)&ZT[m0 + lln][k0 + 8*lhi];
      v8h ahi = *(const v8h*)&ZT[m0 + lln][k0 + 16 + 8*lhi];
      v16h A = SHUF16(alo, ahi);
      v8h blo = *(const v8h*)&DD[lln][k0 + 8*lhi];
      v8h bhi = *(const v8h*)&DD[lln][k0 + 16 + 8*lhi];
      v16h B = SHUF16(blo, bhi);
      acc = WMMA(A, B, acc);
    }
    v8h t;
    #pragma unroll
    for (int r = 0; r < 8; ++r) t[r] = (h16)acc[r];
    *(v8h*)&OvT[lln][m0 + 8*lhi] = t;               // B-layout for pass 5
  }

  // ---- Pass 5: horizontal down-FIR. OutT[q,p] = sum_dzx DD[q][dzx] * OvT[p][dzx] ----
  v8f acc = {};
  #pragma unroll
  for (int k0 = 0; k0 < KZ; k0 += 32) {
    v8h alo = *(const v8h*)&DD[lln][k0 + 8*lhi];
    v8h ahi = *(const v8h*)&DD[lln][k0 + 16 + 8*lhi];
    v16h A = SHUF16(alo, ahi);
    v8h blo = *(const v8h*)&OvT[lln][k0 + 8*lhi];
    v8h bhi = *(const v8h*)&OvT[lln][k0 + 16 + 8*lhi];
    v16h B = SHUF16(blo, bhi);
    acc = WMMA(A, B, acc);
  }

  // Fragment lane holds Out[row = oy0+lln, cols = ox0 + 8*lhi + (0..7)] -> two 16B stores
  float* op = out + (((size_t)(b*CC + c) * HH) + oy0 + lln) * WW + ox0 + 8*lhi;
  v4f s0 = { acc[0], acc[1], acc[2], acc[3] };
  v4f s1 = { acc[4], acc[5], acc[6], acc[7] };
  *(v4f*)op       = s0;
  *(v4f*)(op + 4) = s1;
}

extern "C" void kernel_launch(void* const* d_in, const int* in_sizes, int n_in,
                              void* d_out, int out_size, void* d_ws, size_t ws_size,
                              hipStream_t stream) {
  (void)in_sizes; (void)n_in; (void)out_size; (void)d_ws; (void)ws_size;
  const float* x    = (const float*)d_in[0];
  const float* bias = (const float*)d_in[1];
  const float* upf  = (const float*)d_in[2];
  const float* dnf  = (const float*)d_in[3];
  float* out = (float*)d_out;
  dim3 grid(64 /*8x8 tiles*/, CC, 8);
  flrelu_fused_wmma<<<grid, 32, 0, stream>>>(x, bias, upf, dnf, out);
}
